// Plain_VQ_58703613001740
// MI455X (gfx1250) — compile-verified, matrix operand order checked by reading, not run on hardware
//
#include <hip/hip_runtime.h>
#include <hip/hip_bf16.h>
#include <math.h>

// ---- CDNA5 (gfx1250) wave32 WMMA types ----
typedef __bf16 bf16_t;
typedef __attribute__((ext_vector_type(16))) __bf16 v16bf;
typedef __attribute__((ext_vector_type(8)))  __bf16 v8bf;
typedef __attribute__((ext_vector_type(8)))  float  v8f;
typedef __attribute__((ext_vector_type(4)))  float  v4f;

#define N_ROWS 65536
#define K_CB   1024
#define D_DIM  256

__device__ __forceinline__ unsigned f2u(float f) { return __builtin_bit_cast(unsigned, f); }
// truncate-to-bf16 (upper 16 bits) -> residual split is exact in f32
__device__ __forceinline__ bf16_t bf_trunc(float f) {
    return __builtin_bit_cast(bf16_t, (unsigned short)(f2u(f) >> 16));
}
__device__ __forceinline__ float bf_hi_f(float f) {
    return __builtin_bit_cast(float, f2u(f) & 0xFFFF0000u);
}

// ---------------------------------------------------------------------------
// Kernel 1: codebooks -> bf16 hi/lo split, exact c_sq, zero counts
// ---------------------------------------------------------------------------
__global__ void vq_prep(const float* __restrict__ cb,
                        bf16_t* __restrict__ cb_hi, bf16_t* __restrict__ cb_lo,
                        float* __restrict__ c_sq, int* __restrict__ counts) {
    __shared__ float red[256];
    const int n = blockIdx.x, d = threadIdx.x;
    const float v = cb[(size_t)n * D_DIM + d];
    cb_hi[(size_t)n * D_DIM + d] = bf_trunc(v);
    cb_lo[(size_t)n * D_DIM + d] = bf_trunc(v - bf_hi_f(v));
    red[d] = v * v;
    __syncthreads();
    for (int s = 128; s > 0; s >>= 1) { if (d < s) red[d] += red[d + s]; __syncthreads(); }
    if (d == 0) { c_sq[n] = red[0]; counts[n] = 0; }
}

// ---------------------------------------------------------------------------
// Kernel 2: distance GEMM via v_wmma_f32_16x16x32_bf16 + fused argmin.
// 256 threads = 8 waves; each wave keeps TWO resident 16-row A tiles (32 rows)
// so every streamed B fragment feeds 6 WMMAs (0.67 b128 loads per WMMA).
// ---------------------------------------------------------------------------
__global__ void __launch_bounds__(256, 1)
vq_main(const float* __restrict__ X,
        const bf16_t* __restrict__ cb_hi, const bf16_t* __restrict__ cb_lo,
        const float* __restrict__ c_sq, int* __restrict__ idxs) {
    const int lane = threadIdx.x & 31;
    const int wid  = threadIdx.x >> 5;
    const int half = lane >> 4;       // ISA: lanes 0-15 = K-base 0, lanes 16-31 = K-base 8
    const int mrow = lane & 15;       // A: row M ; B/C: column N
    const int row_base = blockIdx.x * 256 + wid * 32;   // 32 rows per wave

    // ---- Load 2 A tiles (32 x 256 f32) straight into WMMA A-fragment layout.
    // 16-bit A 16x32: v16bf elems 0..7 = K base+0..7, elems 8..15 = K base+16..23.
    v16bf a_hi[2][8], a_lo[2][8];
    float xsq[2][8];
#pragma unroll
    for (int mt = 0; mt < 2; ++mt) {
        float ssq = 0.f;
#pragma unroll
        for (int kc = 0; kc < 8; ++kc) {
            const int k0 = kc * 32 + half * 8;
            const v4f* pa = (const v4f*)(X + (size_t)(row_base + mt * 16 + mrow) * D_DIM + k0);
            v4f q0 = pa[0], q1 = pa[1], q2 = pa[4], q3 = pa[5];   // K+0..7, K+16..23
            float t[16];
#pragma unroll
            for (int e = 0; e < 4; ++e) { t[e] = q0[e]; t[4 + e] = q1[e]; t[8 + e] = q2[e]; t[12 + e] = q3[e]; }
#pragma unroll
            for (int e = 0; e < 16; ++e) {
                const float v = t[e];
                ssq += v * v;
                a_hi[mt][kc][e] = bf_trunc(v);
                a_lo[mt][kc][e] = bf_trunc(v - bf_hi_f(v));
            }
        }
        ssq += __shfl_xor(ssq, 16);            // combine the two K-halves of this row
#pragma unroll
        for (int i = 0; i < 8; ++i) xsq[mt][i] = __shfl(ssq, i + 8 * half);
    }

    float minv[2][8]; int mini[2][8];
#pragma unroll
    for (int mt = 0; mt < 2; ++mt)
#pragma unroll
        for (int i = 0; i < 8; ++i) { minv[mt][i] = 3.4e38f; mini[mt][i] = 0; }

    for (int t = 0; t < 64; ++t) {
        const int n = t * 16 + mrow;           // this lane's codebook column
        const float csq = c_sq[n];
        const bf16_t* bh_base = cb_hi + (size_t)n * D_DIM;
        const bf16_t* bl_base = cb_lo + (size_t)n * D_DIM;
        if (t + 1 < 64) {                       // gfx1250 global_prefetch_b8
            __builtin_prefetch(cb_hi + (size_t)(n + 16) * D_DIM, 0, 3);
            __builtin_prefetch(cb_lo + (size_t)(n + 16) * D_DIM, 0, 3);
        }
        v8f acc0 = {}, acc0c = {}, acc1 = {}, acc1c = {};   // 4 independent XDL chains
#pragma unroll
        for (int kc = 0; kc < 8; ++kc) {
            const int k0 = kc * 32 + half * 8;
            const v8bf* ph = (const v8bf*)(bh_base + k0);
            const v8bf* pl = (const v8bf*)(bl_base + k0);
            v8bf h0 = ph[0], h1 = ph[2], l0 = pl[0], l1 = pl[2];   // 16B vector loads
            v16bf bh, bl;
#pragma unroll
            for (int e = 0; e < 8; ++e) { bh[e] = h0[e]; bh[8 + e] = h1[e]; bl[e] = l0[e]; bl[8 + e] = l1[e]; }
            // x.cT ~= hi*hi + hi*lo + lo*hi  for both M-tiles off one B fetch
            acc0  = __builtin_amdgcn_wmma_f32_16x16x32_bf16(false, a_hi[0][kc], false, bh, (short)0, acc0,  false, false);
            acc1  = __builtin_amdgcn_wmma_f32_16x16x32_bf16(false, a_hi[1][kc], false, bh, (short)0, acc1,  false, false);
            acc0c = __builtin_amdgcn_wmma_f32_16x16x32_bf16(false, a_hi[0][kc], false, bl, (short)0, acc0c, false, false);
            acc1c = __builtin_amdgcn_wmma_f32_16x16x32_bf16(false, a_hi[1][kc], false, bl, (short)0, acc1c, false, false);
            acc0c = __builtin_amdgcn_wmma_f32_16x16x32_bf16(false, a_lo[0][kc], false, bh, (short)0, acc0c, false, false);
            acc1c = __builtin_amdgcn_wmma_f32_16x16x32_bf16(false, a_lo[1][kc], false, bh, (short)0, acc1c, false, false);
        }
#pragma unroll
        for (int i = 0; i < 8; ++i) {
            const float d0 = xsq[0][i] + csq - 2.0f * (acc0[i] + acc0c[i]);
            if (d0 < minv[0][i]) { minv[0][i] = d0; mini[0][i] = n; }
            const float d1 = xsq[1][i] + csq - 2.0f * (acc1[i] + acc1c[i]);
            if (d1 < minv[1][i]) { minv[1][i] = d1; mini[1][i] = n; }
        }
    }

    // Argmin across the 16 lanes of each half (each lane held one column set).
#pragma unroll
    for (int mt = 0; mt < 2; ++mt)
#pragma unroll
        for (int i = 0; i < 8; ++i) {
#pragma unroll
            for (int m = 1; m < 16; m <<= 1) {
                const float ov = __shfl_xor(minv[mt][i], m);
                const int   oi = __shfl_xor(mini[mt][i], m);
                if (ov < minv[mt][i] || (ov == minv[mt][i] && oi < mini[mt][i])) {
                    minv[mt][i] = ov; mini[mt][i] = oi;
                }
            }
            if (mrow == 0) idxs[row_base + mt * 16 + 8 * half + i] = mini[mt][i];
        }
}

// ---------------------------------------------------------------------------
// Kernel 3: gather quantized rows, indices, counts (int atomics), loss partials
// ---------------------------------------------------------------------------
__global__ void vq_finalize(const float* __restrict__ X, const float* __restrict__ cb,
                            const int* __restrict__ idxs, float* __restrict__ outq,
                            float* __restrict__ outidx, int* __restrict__ counts,
                            float* __restrict__ partials) {
    __shared__ float red[256];
    const int tid = threadIdx.x;
    float lsum = 0.f;
#pragma unroll
    for (int r = 0; r < 8; ++r) {
        const int row = blockIdx.x * 8 + r;
        const int idx = idxs[row];
        const float cv = cb[(size_t)idx * D_DIM + tid];
        outq[(size_t)row * D_DIM + tid] = cv;
        const float diff = X[(size_t)row * D_DIM + tid] - cv;
        lsum += diff * diff;
        if (tid == 0) { atomicAdd(&counts[idx], 1); outidx[row] = (float)idx; }
    }
    red[tid] = lsum;
    __syncthreads();
    for (int s = 128; s > 0; s >>= 1) { if (tid < s) red[tid] += red[tid + s]; __syncthreads(); }
    if (tid == 0) partials[blockIdx.x] = red[0];
}

// ---------------------------------------------------------------------------
// Kernel 4: deterministic scalar outputs (loss, perplexity)
// ---------------------------------------------------------------------------
__global__ void vq_scalars(const float* __restrict__ partials, const int* __restrict__ counts,
                           float* __restrict__ out_loss, float* __restrict__ out_perp) {
    __shared__ float red[256];
    const int tid = threadIdx.x;
    float s = 0.f;
    for (int j = 0; j < 32; ++j) s += partials[tid + 256 * j];   // 8192 partials, fixed order
    red[tid] = s;
    __syncthreads();
    for (int k = 128; k > 0; k >>= 1) { if (tid < k) red[tid] += red[tid + k]; __syncthreads(); }
    const float loss = red[0] / (float)((size_t)N_ROWS * D_DIM);
    __syncthreads();
    float e = 0.f;
    for (int j = 0; j < 4; ++j) {
        const float p = (float)counts[tid + 256 * j] / (float)N_ROWS;
        e += p * logf(p + 1e-10f);
    }
    red[tid] = e;
    __syncthreads();
    for (int k = 128; k > 0; k >>= 1) { if (tid < k) red[tid] += red[tid + k]; __syncthreads(); }
    if (tid == 0) { *out_loss = loss; *out_perp = expf(-red[0]); }
}

// ---------------------------------------------------------------------------
extern "C" void kernel_launch(void* const* d_in, const int* in_sizes, int n_in,
                              void* d_out, int out_size, void* d_ws, size_t ws_size,
                              hipStream_t stream) {
    (void)in_sizes; (void)n_in; (void)out_size; (void)ws_size;
    const float* X  = (const float*)d_in[0];   // [N, D] f32
    const float* CB = (const float*)d_in[1];   // [K, D] f32

    float* out      = (float*)d_out;
    float* outq     = out;                               // [N*D] quantized
    float* out_loss = out + (size_t)N_ROWS * D_DIM;      // scalar
    float* out_perp = out_loss + 1;                      // scalar
    float* out_idx  = out_perp + 1;                      // [N] indices (as float)

    char* ws = (char*)d_ws;
    bf16_t* cb_hi    = (bf16_t*)(ws);                    // 512 KB
    bf16_t* cb_lo    = (bf16_t*)(ws + 524288);           // 512 KB
    float*  c_sq     = (float*) (ws + 1048576);          // 4 KB
    int*    counts   = (int*)   (ws + 1052672);          // 4 KB
    int*    idxs     = (int*)   (ws + 1056768);          // 256 KB
    float*  partials = (float*) (ws + 1318912);          // 32 KB

    vq_prep    <<<K_CB,          256, 0, stream>>>(CB, cb_hi, cb_lo, c_sq, counts);
    vq_main    <<<N_ROWS / 256,  256, 0, stream>>>(X, cb_hi, cb_lo, c_sq, idxs);
    vq_finalize<<<N_ROWS / 8,    256, 0, stream>>>(X, CB, idxs, outq, out_idx, counts, partials);
    vq_scalars <<<1,             256, 0, stream>>>(partials, counts, out_loss, out_perp);
}
